// Encoder_cf_31336081391873
// MI455X (gfx1250) — matrix operand order repaired and checked by reading, not compiled, not run on hardware
//
#include <hip/hip_runtime.h>
#include <hip/hip_bf16.h>

#define T_LEN   192
#define B_SZ    128
#define S_SEG   7
#define L_SEG   64
#define SL      (S_SEG * L_SEG)
#define MAXPAD  192
#define ENC_C   512
#define ENC_F   256

typedef __attribute__((ext_vector_type(16))) __bf16 v16bf;
typedef __attribute__((ext_vector_type(8)))  __bf16 v8bf;
typedef __attribute__((ext_vector_type(8)))  float  v8f;

__device__ __forceinline__ __bf16 f2bf(float f) {
    union { float f; unsigned u; } a; a.f = f;
    unsigned r = a.u + 0x7FFFu + ((a.u >> 16) & 1u);   // round-to-nearest-even
    unsigned short hi = (unsigned short)(r >> 16);
    __bf16 out;
    __builtin_memcpy(&out, &hi, 2);
    return out;
}
__device__ __forceinline__ float bf2f(__bf16 h) {
    unsigned short u; __builtin_memcpy(&u, &h, 2);
    union { unsigned u32; float f; } a; a.u32 = ((unsigned)u) << 16;
    return a.f;
}
__device__ __forceinline__ float sigm(float x) { return 1.0f / (1.0f + __expf(-x)); }

// ---------------------------------------------------------------------------
// Weight pack: w[Cout][Cin][5] fp32 -> wpk[tap][Cout][Cpad] bf16 (zero padded)
// ---------------------------------------------------------------------------
__global__ void pack_weights_kernel(const float* __restrict__ w, __bf16* __restrict__ wpk,
                                    int Cin, int Cout, int Cpad)
{
    int idx = blockIdx.x * blockDim.x + threadIdx.x;
    int total = 5 * Cout * Cpad;
    if (idx >= total) return;
    int ci  = idx % Cpad;
    int co  = (idx / Cpad) % Cout;
    int tap = idx / (Cpad * Cout);
    float v = (ci < Cin) ? w[((size_t)co * Cin + ci) * 5 + tap] : 0.0f;
    wpk[idx] = f2bf(v);
}

// ---------------------------------------------------------------------------
// Activation pack (layer 0): fp32 [b][*][T] (batch stride) -> bf16 [b][Cpad][T]
// ---------------------------------------------------------------------------
__global__ void pack_act_kernel(const float* __restrict__ in, __bf16* __restrict__ out,
                                int Cin, int Cpad, long long in_bstride)
{
    int idx = blockIdx.x * blockDim.x + threadIdx.x;
    int total = B_SZ * Cpad * T_LEN;
    if (idx >= total) return;
    int t = idx % T_LEN;
    int c = (idx / T_LEN) % Cpad;
    int b = idx / (T_LEN * Cpad);
    float v = (c < Cin) ? in[(size_t)b * in_bstride + (size_t)c * T_LEN + t] : 0.0f;
    out[idx] = f2bf(v);
}

// ---------------------------------------------------------------------------
// Conv1d (k=5, pad=2) as GEMM, bf16 WMMA 16x16x32.
// A (weights): loaded per-lane directly from packed global bf16 (2x b128).
// B (activations): one extended 36x32 LDS tile per 32-channel step, reused
// by all 5 taps; fragments are 2x ds_load_b128 per lane.
// Block: 256 thr = 8 waves (4 co-tiles x 2 t-tiles) -> tile 64co x 32t.
// ---------------------------------------------------------------------------
__global__ void conv1d_wmma_kernel(const __bf16* __restrict__ act,  // [B][Cpad][T]
                                   long long act_bstride,
                                   const __bf16* __restrict__ wpk,  // [5][Cout][Cpad]
                                   const float* __restrict__ bias,  // [Cout]
                                   float* __restrict__ out,         // [B][Cout][T]
                                   int Cout, int Cpad)
{
    const int b   = blockIdx.z;
    const int co0 = blockIdx.x * 64;
    const int t0  = blockIdx.y * 32;
    const int tid = threadIdx.x;
    const int lane = tid & 31;
    const int wave = tid >> 5;
    const int wco = wave >> 1;   // 0..3
    const int wt  = wave & 1;    // 0..1

    __shared__ __bf16 Bs[36][40];   // [t0-2 .. t0+33][ci]  (padded row stride)

    v8f acc = {};
    const int kciSteps = Cpad >> 5;
    const __bf16* actb = act + (size_t)b * (size_t)act_bstride;

    const int half = lane >> 4;     // 0/1 : selects Kbase (A: 8*half, B: 8*half)
    const int mrow = lane & 15;
    // per-lane base row pointer into packed weights for this wave's co-tile
    const int coA = co0 + wco * 16 + mrow;

    for (int cs = 0; cs < kciSteps; ++cs) {
        const int ci0 = cs << 5;
        __syncthreads();
        // stage extended input tile: Bs[tt][ci] = act[ci0+ci][t0+tt-2]
        for (int i = tid; i < 36 * 32; i += 256) {
            int tt = i >> 5, ci = i & 31;
            int t = t0 + tt - 2;
            __bf16 v = (__bf16)0.0f;
            if (t >= 0 && t < T_LEN)
                v = actb[(size_t)(ci0 + ci) * T_LEN + t];
            Bs[tt][ci] = v;
        }
        // prefetch next channel chunk of activations
        {
            int ci_nxt = (cs + 1 < kciSteps) ? (ci0 + 32) : 0;
            __builtin_prefetch(&actb[(size_t)ci_nxt * T_LEN + t0], 0, 1);
        }
        __syncthreads();

        #pragma unroll
        for (int tap = 0; tap < 5; ++tap) {
            // A fragment: lane -> row M=coA, K = 8*half + e (e<8), +16 (e>=8)
            const __bf16* arow = wpk + ((size_t)tap * Cout + coA) * Cpad + ci0 + half * 8;
            v8bf alo = *(const v8bf*)(arow);
            v8bf ahi = *(const v8bf*)(arow + 16);
            v16bf af;
            #pragma unroll
            for (int e = 0; e < 8; ++e) { af[e] = alo[e]; af[8 + e] = ahi[e]; }
            // B fragment: lane -> col N=mrow, K = 8*half + e (e<8), +16 (e>=8)
            const __bf16* brow = &Bs[tap + wt * 16 + mrow][half * 8];
            v8bf blo = *(const v8bf*)(brow);
            v8bf bhi = *(const v8bf*)(brow + 16);
            v16bf bfv;
            #pragma unroll
            for (int e = 0; e < 8; ++e) { bfv[e] = blo[e]; bfv[8 + e] = bhi[e]; }

            acc = __builtin_amdgcn_wmma_f32_16x16x32_bf16(false, af, false, bfv,
                                                          (short)0, acc, false, false);
        }
    }

    // D layout: lane -> N=lane&15, VGPR r -> M = r + (lane>=16 ? 8 : 0)
    const int n = lane & 15;
    const int mbase = (lane >> 4) * 8;
    const int t = t0 + wt * 16 + n;
    #pragma unroll
    for (int r = 0; r < 8; ++r) {
        int co = co0 + wco * 16 + mbase + r;
        out[((size_t)b * Cout + co) * T_LEN + t] = acc[r] + bias[co];
    }
}

// ---------------------------------------------------------------------------
// GroupNorm (+ReLU), one block per (group, batch). C/groups == 16 channels.
// ---------------------------------------------------------------------------
__global__ void gn_relu_kernel(const float* __restrict__ in, float* __restrict__ out,
                               const float* __restrict__ gamma, const float* __restrict__ beta,
                               int C, int groups)
{
    const int g = blockIdx.x, b = blockIdx.y;
    const int cpg = C / groups;
    const size_t base = ((size_t)b * C + (size_t)g * cpg) * T_LEN;
    const int n = cpg * T_LEN;

    float s = 0.f, s2 = 0.f;
    for (int i = threadIdx.x; i < n; i += blockDim.x) {
        float v = in[base + i]; s += v; s2 += v * v;
    }
    __shared__ float rs[8], rs2[8];
    for (int off = 16; off; off >>= 1) { s += __shfl_down(s, off); s2 += __shfl_down(s2, off); }
    int wave = threadIdx.x >> 5, lane = threadIdx.x & 31;
    if (lane == 0) { rs[wave] = s; rs2[wave] = s2; }
    __syncthreads();
    if (wave == 0) {
        s  = (lane < 8) ? rs[lane]  : 0.f;
        s2 = (lane < 8) ? rs2[lane] : 0.f;
        for (int off = 4; off; off >>= 1) { s += __shfl_down(s, off); s2 += __shfl_down(s2, off); }
        if (lane == 0) {
            float m = s / n;
            rs[0] = m;
            rs2[0] = s2 / n - m * m;
        }
    }
    __syncthreads();
    const float mean = rs[0];
    const float inv  = rsqrtf(rs2[0] + 1e-5f);
    for (int i = threadIdx.x; i < n; i += blockDim.x) {
        int c = i / T_LEN;
        float v = (in[base + i] - mean) * inv * gamma[g * cpg + c] + beta[g * cpg + c];
        out[base + i] = v > 0.f ? v : 0.f;
    }
}

// ---------------------------------------------------------------------------
// interp_lnr: one block per batch; writes bf16 channel-major outputs that feed
// the next conv layer / the LSTM projections directly.
// ---------------------------------------------------------------------------
__global__ void interp_kernel(const float* __restrict__ xn,  // [B][ENC_C][T] fp32
                              const float* __restrict__ fn,  // [B][ENC_F][T] fp32
                              __bf16* __restrict__ xo,       // [B][ENC_C][T] bf16
                              __bf16* __restrict__ fo,       // [B][ENC_F][T] bf16
                              const float* __restrict__ scales,  // [B*S]
                              const int*   __restrict__ lenseg)  // [B*S]
{
    const int b = blockIdx.x;
    __shared__ int   s_idx[SL];
    __shared__ float s_lam[SL];
    __shared__ unsigned char s_mask[SL];
    __shared__ int   s_pos[SL];
    __shared__ int   s_off[S_SEG];

    if (threadIdx.x == 0) {
        int acc = 0;
        for (int s = 0; s < S_SEG; ++s) { s_off[s] = acc; acc += lenseg[b * S_SEG + s]; }
    }
    __syncthreads();

    for (int j = threadIdx.x; j < SL; j += blockDim.x) {
        int s = j / L_SEG, jj = j % L_SEG;
        float sc = scales[b * S_SEG + s];
        float idx_scaled = (float)jj / sc;
        float fl = floorf(idx_scaled);
        float lam = idx_scaled - fl;
        int len = lenseg[b * S_SEG + s];
        int idx_org = (int)fl + s_off[s];
        bool m = (fl < (float)len - 1.0f) && (idx_org < T_LEN - 1);
        int idxb = idx_org;
        if (idxb < 0) idxb = 0;
        if (idxb > T_LEN - 2) idxb = T_LEN - 2;
        s_idx[j] = idxb; s_lam[j] = lam; s_mask[j] = m ? 1 : 0;
    }
    __syncthreads();
    if (threadIdx.x == 0) {
        int c = 0;
        for (int j = 0; j < SL; ++j) {
            int p = MAXPAD;
            if (s_mask[j]) { if (c < MAXPAD) p = c; c++; }
            s_pos[j] = p;
        }
    }
    __syncthreads();

    const __bf16 z = (__bf16)0.0f;
    for (int i = threadIdx.x; i < ENC_C * T_LEN; i += blockDim.x)
        xo[(size_t)b * ENC_C * T_LEN + i] = z;
    for (int i = threadIdx.x; i < ENC_F * T_LEN; i += blockDim.x)
        fo[(size_t)b * ENC_F * T_LEN + i] = z;
    __threadfence_block();
    __syncthreads();

    for (int j = 0; j < SL; ++j) {
        int p = s_pos[j];
        if (!s_mask[j] || p >= MAXPAD) continue;
        int i0 = s_idx[j];
        float lam = s_lam[j], oml = 1.0f - lam;
        for (int c = threadIdx.x; c < ENC_C; c += blockDim.x) {
            const float* src = xn + ((size_t)b * ENC_C + c) * T_LEN;
            xo[((size_t)b * ENC_C + c) * T_LEN + p] = f2bf(oml * src[i0] + lam * src[i0 + 1]);
        }
        for (int c = threadIdx.x; c < ENC_F; c += blockDim.x) {
            const float* src = fn + ((size_t)b * ENC_F + c) * T_LEN;
            fo[((size_t)b * ENC_F + c) * T_LEN + p] = f2bf(oml * src[i0] + lam * src[i0 + 1]);
        }
    }
}

// ---------------------------------------------------------------------------
// LSTM input projection; channel-major bf16 input ([b][K][T]) variant and
// row-major fp32 input ([b][T][K]) variant.
// ---------------------------------------------------------------------------
__global__ void lstm_pre_cm_kernel(const __bf16* __restrict__ in, const float* __restrict__ wih,
                                   const float* __restrict__ bih, const float* __restrict__ bhh,
                                   float* __restrict__ pre, int K, int NG)
{
    int idx = blockIdx.x * blockDim.x + threadIdx.x;
    int total = B_SZ * T_LEN * NG;
    if (idx >= total) return;
    int g = idx % NG;
    int t = (idx / NG) % T_LEN;
    int b = idx / (NG * T_LEN);
    float s = bih[g] + bhh[g];
    const __bf16* ip = in + (size_t)b * K * T_LEN + t;
    const float* wp = wih + (size_t)g * K;
    for (int k = 0; k < K; ++k) s += bf2f(ip[(size_t)k * T_LEN]) * wp[k];
    pre[idx] = s;
}

__global__ void lstm_pre_rm_kernel(const float* __restrict__ in, const float* __restrict__ wih,
                                   const float* __restrict__ bih, const float* __restrict__ bhh,
                                   float* __restrict__ pre, int K, int NG)
{
    int idx = blockIdx.x * blockDim.x + threadIdx.x;
    int total = B_SZ * T_LEN * NG;
    if (idx >= total) return;
    int g = idx % NG;
    int t = (idx / NG) % T_LEN;
    int b = idx / (NG * T_LEN);
    float s = bih[g] + bhh[g];
    const float* ip = in + ((size_t)b * T_LEN + t) * K;
    const float* wp = wih + (size_t)g * K;
    for (int k = 0; k < K; ++k) s += ip[k] * wp[k];
    pre[idx] = s;
}

// ---------------------------------------------------------------------------
// LSTM recurrence: one wave32 per sequence; h/c one-per-lane; shfl broadcast.
// ---------------------------------------------------------------------------
template <int H, int NG>
__global__ void lstm_scan_kernel(const float* __restrict__ pre,   // [B][T][NG]
                                 const float* __restrict__ whh,   // [NG][H]
                                 float* __restrict__ hs_cat,      // [B][T][stride]
                                 int stride, int col_off, int reverse)
{
    constexpr int GPL = NG / 32;
    __shared__ float s_whh[NG * H];
    for (int i = threadIdx.x; i < NG * H; i += blockDim.x) s_whh[i] = whh[i];
    __syncthreads();

    const int wave = threadIdx.x >> 5, lane = threadIdx.x & 31;
    const int b = blockIdx.x * (blockDim.x >> 5) + wave;
    float h = 0.f, c = 0.f;
    const float* preb = pre + (size_t)b * T_LEN * NG;
    float* outb = hs_cat + (size_t)b * T_LEN * stride + col_off;

    for (int step = 0; step < T_LEN; ++step) {
        int t = reverse ? (T_LEN - 1 - step) : step;
        float gate[GPL];
        #pragma unroll
        for (int q = 0; q < GPL; ++q) gate[q] = preb[(size_t)t * NG + lane + 32 * q];
        #pragma unroll
        for (int hh = 0; hh < H; ++hh) {
            float hv = __shfl(h, hh);
            #pragma unroll
            for (int q = 0; q < GPL; ++q)
                gate[q] += s_whh[(lane + 32 * q) * H + hh] * hv;
        }
        float g4[4];
        #pragma unroll
        for (int k = 0; k < 4; ++k) {
            int src = (H * k + (lane & (H - 1))) & 31;
            g4[k] = __shfl(gate[(H * k) >> 5], src);
        }
        float nc = sigm(g4[1]) * c + sigm(g4[0]) * tanhf(g4[2]);
        c = nc;
        h = sigm(g4[3]) * tanhf(c);
        if (lane < H) outb[(size_t)t * stride + lane] = h;
    }
}

// ---------------------------------------------------------------------------
// codes = concat(hs[:, freq-1::freq, :H], hs[:, ::freq, H:])
// ---------------------------------------------------------------------------
__global__ void codes_kernel(const float* __restrict__ hs, float* __restrict__ out,
                             int twoH, int freq)
{
    const int H = twoH >> 1;
    const int nI = T_LEN / freq;
    int idx = blockIdx.x * blockDim.x + threadIdx.x;
    int total = B_SZ * nI * twoH;
    if (idx >= total) return;
    int j = idx % twoH;
    int i = (idx / twoH) % nI;
    int b = idx / (twoH * nI);
    int t = (j < H) ? (i * freq + freq - 1) : (i * freq);
    out[idx] = hs[((size_t)b * T_LEN + t) * twoH + j];
}

// ---------------------------------------------------------------------------
extern "C" void kernel_launch(void* const* d_in, const int* in_sizes, int n_in,
                              void* d_out, int out_size, void* d_ws, size_t ws_size,
                              hipStream_t stream)
{
    const float* x_f0   = (const float*)d_in[0];
    const float* scales = (const float*)d_in[1];
    const int*   lenseg = (const int*)d_in[2];

    int p = 3;
    const float *cw[3], *cb[3], *cgg[3], *cgb[3];
    for (int i = 0; i < 3; ++i) {
        cw[i]  = (const float*)d_in[p++]; cb[i]  = (const float*)d_in[p++];
        cgg[i] = (const float*)d_in[p++]; cgb[i] = (const float*)d_in[p++];
    }
    const float *fw[3], *fbb[3], *fgg[3], *fgb[3];
    for (int i = 0; i < 3; ++i) {
        fw[i]  = (const float*)d_in[p++]; fbb[i] = (const float*)d_in[p++];
        fgg[i] = (const float*)d_in[p++]; fgb[i] = (const float*)d_in[p++];
    }
    const float *lc_wih[2][2], *lc_whh[2][2], *lc_bih[2][2], *lc_bhh[2][2];
    for (int l = 0; l < 2; ++l)
        for (int d = 0; d < 2; ++d) {
            lc_wih[l][d] = (const float*)d_in[p++];
            lc_whh[l][d] = (const float*)d_in[p++];
            lc_bih[l][d] = (const float*)d_in[p++];
            lc_bhh[l][d] = (const float*)d_in[p++];
        }
    const float *lf_wih[2], *lf_whh[2], *lf_bih[2], *lf_bhh[2];
    for (int d = 0; d < 2; ++d) {
        lf_wih[d] = (const float*)d_in[p++];
        lf_whh[d] = (const float*)d_in[p++];
        lf_bih[d] = (const float*)d_in[p++];
        lf_bhh[d] = (const float*)d_in[p++];
    }

    // ---- workspace carve (bytes) ----
    char* wsb = (char*)d_ws;
    float* cbuf1 = (float*)wsb;                                   // fp32 conv pre-act C: B*512*T
    wsb += (size_t)B_SZ * ENC_C * T_LEN * 4;
    float* fbuf1 = (float*)wsb;                                   // fp32 conv pre-act F: B*256*T
    wsb += (size_t)B_SZ * ENC_F * T_LEN * 4;
    __bf16* abufC = (__bf16*)wsb;                                 // bf16 activations C (max 512 rows)
    wsb += (size_t)B_SZ * ENC_C * T_LEN * 2;
    __bf16* abufF = (__bf16*)wsb;                                 // bf16 activations F (max 288 rows)
    wsb += (size_t)B_SZ * 288 * T_LEN * 2;
    __bf16* wpk = (__bf16*)wsb;                                   // packed weights (max 5*512*512)
    wsb += (size_t)5 * ENC_C * ENC_C * 2;

    // ---- layer 0 input packing ----
    {
        int tot = B_SZ * 96 * T_LEN;
        pack_act_kernel<<<(tot + 255) / 256, 256, 0, stream>>>(x_f0, abufC, 80, 96, 337LL * T_LEN);
        tot = B_SZ * 288 * T_LEN;
        pack_act_kernel<<<(tot + 255) / 256, 256, 0, stream>>>(x_f0 + 80LL * T_LEN, abufF, 257, 288, 337LL * T_LEN);
    }

    for (int i = 0; i < 3; ++i) {
        // -------- C path --------
        int cinC  = (i == 0) ? 80  : ENC_C;
        int cpadC = (i == 0) ? 96  : ENC_C;
        {
            int tot = 5 * ENC_C * cpadC;
            pack_weights_kernel<<<(tot + 255) / 256, 256, 0, stream>>>(cw[i], wpk, cinC, ENC_C, cpadC);
        }
        conv1d_wmma_kernel<<<dim3(ENC_C / 64, T_LEN / 32, B_SZ), 256, 0, stream>>>(
            abufC, (long long)cpadC * T_LEN, wpk, cb[i], cbuf1, ENC_C, cpadC);
        gn_relu_kernel<<<dim3(32, B_SZ), 256, 0, stream>>>(cbuf1, cbuf1, cgg[i], cgb[i], ENC_C, 32);

        // -------- F path --------
        int cinF  = (i == 0) ? 257 : ENC_F;
        int cpadF = (i == 0) ? 288 : ENC_F;
        {
            int tot = 5 * ENC_F * cpadF;
            pack_weights_kernel<<<(tot + 255) / 256, 256, 0, stream>>>(fw[i], wpk, cinF, ENC_F, cpadF);
        }
        conv1d_wmma_kernel<<<dim3(ENC_F / 64, T_LEN / 32, B_SZ), 256, 0, stream>>>(
            abufF, (long long)cpadF * T_LEN, wpk, fbb[i], fbuf1, ENC_F, cpadF);
        gn_relu_kernel<<<dim3(16, B_SZ), 256, 0, stream>>>(fbuf1, fbuf1, fgg[i], fgb[i], ENC_F, 16);

        // -------- interp (writes bf16 for next conv / LSTM) --------
        interp_kernel<<<B_SZ, 256, 0, stream>>>(cbuf1, fbuf1, abufC, abufF,
                                                scales + (size_t)i * B_SZ * S_SEG,
                                                lenseg + (size_t)i * B_SZ * S_SEG);
    }

    // LSTM scratch reuses cbuf1 (free after final interp): 4 slots of 3,145,728 floats
    const size_t slot = 3145728;
    float* preA = cbuf1;
    float* preB = cbuf1 + slot;
    float* hs1  = cbuf1 + 2 * slot;
    float* hs2  = cbuf1 + 3 * slot;
    float* outp = (float*)d_out;

    // ---- C path: 2-layer biLSTM, H=8 ----
    {
        int tot = B_SZ * T_LEN * 32;
        int nb = (tot + 255) / 256;
        lstm_pre_cm_kernel<<<nb, 256, 0, stream>>>(abufC, lc_wih[0][0], lc_bih[0][0], lc_bhh[0][0], preA, ENC_C, 32);
        lstm_pre_cm_kernel<<<nb, 256, 0, stream>>>(abufC, lc_wih[0][1], lc_bih[0][1], lc_bhh[0][1], preB, ENC_C, 32);
        lstm_scan_kernel<8, 32><<<16, 256, 0, stream>>>(preA, lc_whh[0][0], hs1, 16, 0, 0);
        lstm_scan_kernel<8, 32><<<16, 256, 0, stream>>>(preB, lc_whh[0][1], hs1, 16, 8, 1);
        lstm_pre_rm_kernel<<<nb, 256, 0, stream>>>(hs1, lc_wih[1][0], lc_bih[1][0], lc_bhh[1][0], preA, 16, 32);
        lstm_pre_rm_kernel<<<nb, 256, 0, stream>>>(hs1, lc_wih[1][1], lc_bih[1][1], lc_bhh[1][1], preB, 16, 32);
        lstm_scan_kernel<8, 32><<<16, 256, 0, stream>>>(preA, lc_whh[1][0], hs2, 16, 0, 0);
        lstm_scan_kernel<8, 32><<<16, 256, 0, stream>>>(preB, lc_whh[1][1], hs2, 16, 8, 1);
        int ctot = B_SZ * (T_LEN / 8) * 16;
        codes_kernel<<<(ctot + 255) / 256, 256, 0, stream>>>(hs2, outp, 16, 8);
    }

    // ---- F path: 1-layer biLSTM, H=32 ----
    {
        int tot = B_SZ * T_LEN * 128;
        int nb = (tot + 255) / 256;
        lstm_pre_cm_kernel<<<nb, 256, 0, stream>>>(abufF, lf_wih[0], lf_bih[0], lf_bhh[0], preA, ENC_F, 128);
        lstm_pre_cm_kernel<<<nb, 256, 0, stream>>>(abufF, lf_wih[1], lf_bih[1], lf_bhh[1], preB, ENC_F, 128);
        lstm_scan_kernel<32, 128><<<16, 256, 0, stream>>>(preA, lf_whh[0], hs1, 64, 0, 0);
        lstm_scan_kernel<32, 128><<<16, 256, 0, stream>>>(preB, lf_whh[1], hs1, 64, 32, 1);
        int ftot = B_SZ * (T_LEN / 8) * 64;
        codes_kernel<<<(ftot + 255) / 256, 256, 0, stream>>>(hs1, outp + (size_t)B_SZ * (T_LEN / 8) * 16, 64, 8);
    }
}
